// GatedAttention_42356967473584
// MI455X (gfx1250) — matrix operand order repaired and checked by reading, not compiled, not run on hardware
//
#include <hip/hip_runtime.h>
#include <hip/hip_bf16.h>

// ---------------------------------------------------------------------------
// Gated attention, fused flash-attention style, bf16 WMMA (gfx1250 / CDNA5)
// v3: double-buffered async global->LDS staging (ASYNCcnt path) in k_attn
// ---------------------------------------------------------------------------

typedef __attribute__((ext_vector_type(16))) __bf16        v16bf;
typedef __attribute__((ext_vector_type(8)))  float         v8f;
typedef __attribute__((ext_vector_type(4)))  unsigned int  u32x4;
typedef __attribute__((ext_vector_type(4)))  float         f32x4;

#define B_  4
#define LD_ 4096
#define LM_ 4096
#define D_  256
#define H_  256
#define O_  256

union Frag {
  v16bf          bf;
  unsigned short us[16];
  u32x4          u4[2];
};

union Pack8 {
  u32x4          v;
  unsigned short us[8];
};

__device__ inline unsigned short f2bf(float x) {
  // round-to-nearest-even fp32 -> bf16
  unsigned int u = __float_as_uint(x);
  unsigned int r = u + 0x7fffu + ((u >> 16) & 1u);
  return (unsigned short)(r >> 16);
}

// butterfly reductions within each 16-lane half (xor 1,2,4,8 stay in-half)
__device__ inline float halfmax(float x) {
  x = fmaxf(x, __shfl_xor(x, 1, 32));
  x = fmaxf(x, __shfl_xor(x, 2, 32));
  x = fmaxf(x, __shfl_xor(x, 4, 32));
  x = fmaxf(x, __shfl_xor(x, 8, 32));
  return x;
}
__device__ inline float halfsum(float x) {
  x += __shfl_xor(x, 1, 32);
  x += __shfl_xor(x, 2, 32);
  x += __shfl_xor(x, 4, 32);
  x += __shfl_xor(x, 8, 32);
  return x;
}

// async 16-byte global -> LDS copy (per lane), tracked by ASYNCcnt
__device__ inline void async_cp_b128(unsigned lds, const void* g) {
  asm volatile("global_load_async_to_lds_b128 %0, %1, off"
               :: "v"(lds), "v"((unsigned long long)(size_t)g)
               : "memory");
}

// ---------------------------------------------------------------------------
// Kernel 1: one-time fp32 -> bf16 conversions.
//   Xb  = bf16(input)   [B*Ld, D]  row-major (vectorized 8-wide)
//   Mb  = bf16(memory)  [B*Lm, D]  row-major (vectorized 8-wide)
//   Vt  = bf16(memory)  [B, D, Lm] transposed (for PV B-fragments)
//   Wb_in/Wb_mem [H, D], Wb2 [O, 2D] bf16 weights
// ---------------------------------------------------------------------------
__global__ void k_convert(const float* __restrict__ input,
                          const float* __restrict__ memory,
                          const float* __restrict__ Win,
                          const float* __restrict__ Wmem,
                          const float* __restrict__ W2,
                          unsigned short* __restrict__ Xb,
                          unsigned short* __restrict__ Mb,
                          unsigned short* __restrict__ Vt,
                          unsigned short* __restrict__ Wb_in,
                          unsigned short* __restrict__ Wb_mem,
                          unsigned short* __restrict__ Wb2) {
  int idx = blockIdx.x * blockDim.x + threadIdx.x;
  int stride = gridDim.x * blockDim.x;

  const int n8 = (B_ * LD_ * D_) / 8;          // 8 elements per thread-iter
  for (int i = idx; i < n8; i += stride) {
    const f32x4* s = reinterpret_cast<const f32x4*>(input) + (size_t)i * 2;
    f32x4 a = s[0], c = s[1];
    Pack8 w;
    w.us[0] = f2bf(a.x); w.us[1] = f2bf(a.y); w.us[2] = f2bf(a.z); w.us[3] = f2bf(a.w);
    w.us[4] = f2bf(c.x); w.us[5] = f2bf(c.y); w.us[6] = f2bf(c.z); w.us[7] = f2bf(c.w);
    reinterpret_cast<u32x4*>(Xb)[i] = w.v;
  }
  const int m8 = (B_ * LM_ * D_) / 8;
  for (int i = idx; i < m8; i += stride) {
    const f32x4* s = reinterpret_cast<const f32x4*>(memory) + (size_t)i * 2;
    f32x4 a = s[0], c = s[1];
    Pack8 w;
    w.us[0] = f2bf(a.x); w.us[1] = f2bf(a.y); w.us[2] = f2bf(a.z); w.us[3] = f2bf(a.w);
    w.us[4] = f2bf(c.x); w.us[5] = f2bf(c.y); w.us[6] = f2bf(c.z); w.us[7] = f2bf(c.w);
    reinterpret_cast<u32x4*>(Mb)[i] = w.v;
  }
  // transposed copy of memory (scattered 2B writes; one-time, L2-resident)
  const int NM = B_ * LM_ * D_;
  for (int j = idx; j < NM; j += stride) {
    int b = j / (LM_ * D_);
    int r = j - b * (LM_ * D_);
    int m = r / D_;
    int d = r - m * D_;
    Vt[((size_t)b * D_ + d) * LM_ + m] = f2bf(memory[j]);
  }
  // weights (small)
  for (int i = idx; i < H_ * D_; i += stride) Wb_in[i]  = f2bf(Win[i]);
  for (int i = idx; i < H_ * D_; i += stride) Wb_mem[i] = f2bf(Wmem[i]);
  for (int i = idx; i < O_ * 2 * D_; i += stride) Wb2[i] = f2bf(W2[i]);
}

// ---------------------------------------------------------------------------
// Kernel 2: projections  Qd = relu(Xb @ Wb_in^T + bin)   (bf16, [B*Ld, H])
//                        Kd = relu(Mb @ Wb_mem^T + bmem) (bf16, [B*Lm, H])
// one 16x16 output tile per wave, K = 256 in 8 chunks of 32, all-bf16 loads.
// ---------------------------------------------------------------------------
__global__ void __launch_bounds__(256)
k_project(const unsigned short* __restrict__ Xb,
          const unsigned short* __restrict__ Mb,
          const unsigned short* __restrict__ Wb_in, const float* __restrict__ bin,
          const unsigned short* __restrict__ Wb_mem, const float* __restrict__ bmem,
          unsigned short* __restrict__ Qd, unsigned short* __restrict__ Kd) {
  const int lane = threadIdx.x & 31;
  const int wave = threadIdx.x >> 5;
  const int gw = blockIdx.x * 8 + wave;   // 32768 waves
  const int rt = gw >> 4;                 // row tile 0..2047 (combined rows)
  const int ct = gw & 15;                 // col tile 0..15
  const int rowbase = rt * 16;
  const bool isInput = rowbase < B_ * LD_;
  const unsigned short* src = isInput ? Xb : Mb;
  const unsigned short* W   = isInput ? Wb_in : Wb_mem;
  const float* bias = isInput ? bin : bmem;
  unsigned short* dst = isInput ? Qd : Kd;
  const int segbase = isInput ? rowbase : rowbase - B_ * LD_;

  const int ln = lane & 15;
  const int hi = (lane >> 4) & 1;
  const unsigned short* ap = src + (size_t)(segbase + ln) * D_;  // A: lane = M
  const unsigned short* bp = W + (size_t)(ct * 16 + ln) * D_;    // B: lane = N

  v8f acc = {};
#pragma unroll
  for (int kb = 0; kb < 8; ++kb) {
    Frag a, bfr;
    const u32x4* pa = reinterpret_cast<const u32x4*>(ap + kb * 32 + hi * 16);
    a.u4[0] = pa[0]; a.u4[1] = pa[1];      // 16 consecutive K
    const u32x4* pb = reinterpret_cast<const u32x4*>(bp + kb * 32 + hi * 16);
    bfr.u4[0] = pb[0]; bfr.u4[1] = pb[1];  // 16 consecutive K
    acc = __builtin_amdgcn_wmma_f32_16x16x32_bf16(false, a.bf, false, bfr.bf,
                                                  (short)0, acc, false, false);
  }
  const int col = ct * 16 + ln;
  const float bv = bias[col];
#pragma unroll
  for (int v = 0; v < 8; ++v) {           // C layout: VGPR v = row (within half)
    int row = segbase + v + hi * 8;
    float z = acc[v] + bv;
    dst[(size_t)row * H_ + col] = f2bf(fmaxf(z, 0.0f));
  }
}

// ---------------------------------------------------------------------------
// Kernel 3: fused masked flash-attention with double-buffered ASYNC staging.
// Block = 8 waves, same batch, iterating keys in lockstep. Per 32-key chunk:
//   global_load_async_to_lds_b128 stages K (16KB) and V (8KB) into the
//   alternate LDS buffer while WMMA consumes the current one.
// ---------------------------------------------------------------------------
__global__ void __launch_bounds__(256)
k_attn(const unsigned short* __restrict__ Qd,
       const unsigned short* __restrict__ Kd,
       const unsigned short* __restrict__ Vt,
       const float* __restrict__ mask,
       unsigned short* __restrict__ O1) {
  __shared__ unsigned short sK[2][32 * 256];   // [buf][key][d]   2 x 16 KB
  __shared__ unsigned short sV[2][256 * 32];   // [buf][d][key]   2 x 16 KB
  __shared__ unsigned short sP[8 * 16 * 32];   // per-wave P tile 8 KB
  const int lane = threadIdx.x & 31;
  const int wave = threadIdx.x >> 5;
  const int gw = blockIdx.x * 8 + wave;        // query tile 0..1023
  const int b  = gw >> 8;                      // 256 tiles per batch; 8 | 256
  const int qt = gw & 255;
  const int qrowbase = b * LD_ + qt * 16;
  const int ln = lane & 15;
  const int hi = (lane >> 4) & 1;

  const unsigned short* gkbase = Kd + (size_t)b * LM_ * H_;
  const unsigned short* gvbase = Vt + (size_t)b * D_ * LM_;
  const unsigned sKb[2] = {(unsigned)(size_t)(void*)&sK[0][0],
                           (unsigned)(size_t)(void*)&sK[1][0]};
  const unsigned sVb[2] = {(unsigned)(size_t)(void*)&sV[0][0],
                           (unsigned)(size_t)(void*)&sV[1][0]};

  // issue async staging of one 32-key chunk into buffer bsel (6 ops/wave)
  auto stage = [&](int kc, int bsel) {
    const unsigned short* gk = gkbase + (size_t)kc * H_;   // flat 16KB block
#pragma unroll
    for (int r = 0; r < 4; ++r) {
      int i = threadIdx.x + 256 * r;                       // 0..1023 x 16B
      async_cp_b128(sKb[bsel] + i * 16, gk + i * 8);
    }
    const unsigned short* gv = gvbase + kc;                // 32 keys per d-row
#pragma unroll
    for (int r = 0; r < 2; ++r) {
      int i = threadIdx.x + 256 * r;                       // 0..511 x 16B
      int d = i >> 1, half = i & 1;
      async_cp_b128(sVb[bsel] + i * 16, gv + (size_t)d * LM_ + half * 16);
    }
  };

  // cache the 16x256 Q tile as 8 A-fragments (two b128 loads each)
  Frag qf[8];
  {
    const unsigned short* qp = Qd + (size_t)(qrowbase + ln) * H_;
#pragma unroll
    for (int kb = 0; kb < 8; ++kb) {
      const u32x4* p = reinterpret_cast<const u32x4*>(qp + kb * 32 + hi * 16);
      qf[kb].u4[0] = p[0];
      qf[kb].u4[1] = p[1];
    }
  }

  v8f acc[16];
#pragma unroll
  for (int t = 0; t < 16; ++t) acc[t] = (v8f){};
  float mrow[8], lrow[8];
#pragma unroll
  for (int v = 0; v < 8; ++v) { mrow[v] = -3.0e38f; lrow[v] = 0.0f; }

  unsigned short* myP = sP + wave * (16 * 32);
  const float* mrowp = mask + b * LM_;

  stage(0, 0);                                  // prologue: chunk 0 -> buf 0

  for (int kc = 0; kc < LM_; kc += 32) {
    const int cur = (kc >> 5) & 1;
    __syncthreads();          // all waves done reading buffer cur^1
    if (kc + 32 < LM_) {
      stage(kc + 32, cur ^ 1);                  // overlap next chunk's copies
      // async loads complete in order: <=6 outstanding => current chunk done
      asm volatile("s_wait_asynccnt 0x6" ::: "memory");
    } else {
      asm volatile("s_wait_asynccnt 0x0" ::: "memory");
    }
    __syncthreads();          // current chunk visible to all waves

    const unsigned short* sKc = &sK[cur][0];
    const unsigned short* sVc = &sV[cur][0];

    // ---- S = Q K^T, two 16-key tiles (B-fragments from LDS) ----
    v8f s0 = {}, s1 = {};
    const unsigned short* kp0 = sKc + (size_t)ln * H_ + hi * 16;
    const unsigned short* kp1 = kp0 + 16 * H_;
#pragma unroll
    for (int kb = 0; kb < 8; ++kb) {
      Frag k0, k1;
      const u32x4* p0 = reinterpret_cast<const u32x4*>(kp0 + kb * 32);
      k0.u4[0] = p0[0]; k0.u4[1] = p0[1];
      const u32x4* p1 = reinterpret_cast<const u32x4*>(kp1 + kb * 32);
      k1.u4[0] = p1[0]; k1.u4[1] = p1[1];
      s0 = __builtin_amdgcn_wmma_f32_16x16x32_bf16(false, qf[kb].bf, false, k0.bf,
                                                   (short)0, s0, false, false);
      s1 = __builtin_amdgcn_wmma_f32_16x16x32_bf16(false, qf[kb].bf, false, k1.bf,
                                                   (short)0, s1, false, false);
    }
    const float pen0 = 1.0e30f * (1.0f - mrowp[kc + ln]);
    const float pen1 = 1.0e30f * (1.0f - mrowp[kc + 16 + ln]);

    // ---- online softmax (row = VGPR index within half) ----
#pragma unroll
    for (int v = 0; v < 8; ++v) {
      float a0 = s0[v] * 0.0625f - pen0;      // 1/sqrt(256) = 1/16
      float a1 = s1[v] * 0.0625f - pen1;
      float mn = fmaxf(mrow[v], halfmax(fmaxf(a0, a1)));
      float alpha = __expf(mrow[v] - mn);
      float p0 = __expf(a0 - mn);
      float p1 = __expf(a1 - mn);
      lrow[v] = lrow[v] * alpha + halfsum(p0 + p1);
      mrow[v] = mn;
      int prow = v + hi * 8;
      myP[prow * 32 + ln]      = f2bf(p0);
      myP[prow * 32 + 16 + ln] = f2bf(p1);
#pragma unroll
      for (int t = 0; t < 16; ++t) acc[t][v] *= alpha;
    }
    asm volatile("s_wait_dscnt 0x0" ::: "memory");  // same-wave LDS RAW

    // P as A-fragment (16x32) from LDS: two ds_load_b128
    Frag pf;
    {
      const u32x4* p = reinterpret_cast<const u32x4*>(myP + ln * 32 + hi * 16);
      pf.u4[0] = p[0];
      pf.u4[1] = p[1];
    }

    // ---- O += P * V : 16 d-column tiles (B-fragments from LDS) ----
#pragma unroll
    for (int t = 0; t < 16; ++t) {
      Frag vf;
      const u32x4* p =
          reinterpret_cast<const u32x4*>(sVc + (size_t)(t * 16 + ln) * 32 + hi * 16);
      vf.u4[0] = p[0]; vf.u4[1] = p[1];
      acc[t] = __builtin_amdgcn_wmma_f32_16x16x32_bf16(false, pf.bf, false, vf.bf,
                                                       (short)0, acc[t], false, false);
    }
  }

  // epilogue: normalize by l, store bf16 output_one
#pragma unroll
  for (int v = 0; v < 8; ++v) {
    float inv = 1.0f / lrow[v];
    int row = qrowbase + v + hi * 8;
#pragma unroll
    for (int t = 0; t < 16; ++t)
      O1[(size_t)row * D_ + t * 16 + ln] = f2bf(acc[t][v] * inv);
  }
}

// ---------------------------------------------------------------------------
// Kernel 4: z = [input, output_one] @ W2^T + b2 ; out = sigmoid(z)*tanh(z)
// all-bf16 fragment loads (weights pre-converted).
// ---------------------------------------------------------------------------
__global__ void __launch_bounds__(256)
k_glu(const unsigned short* __restrict__ Xb,
      const unsigned short* __restrict__ O1,
      const unsigned short* __restrict__ Wb2, const float* __restrict__ b2,
      float* __restrict__ out) {
  const int lane = threadIdx.x & 31;
  const int wave = threadIdx.x >> 5;
  const int gw = blockIdx.x * 8 + wave;  // 16384 waves
  const int rt = gw >> 4;                // 0..1023
  const int ct = gw & 15;
  const int ln = lane & 15;
  const int hi = (lane >> 4) & 1;
  const unsigned short* axp = Xb + (size_t)(rt * 16 + ln) * D_;
  const unsigned short* aop = O1 + (size_t)(rt * 16 + ln) * D_;
  const unsigned short* bp  = Wb2 + (size_t)(ct * 16 + ln) * (2 * D_);

  v8f acc = {};
#pragma unroll
  for (int kb = 0; kb < 8; ++kb) {       // first half: input * W2[:, :256]
    Frag a, bfr;
    const u32x4* pa = reinterpret_cast<const u32x4*>(axp + kb * 32 + hi * 16);
    a.u4[0] = pa[0]; a.u4[1] = pa[1];
    const u32x4* pb = reinterpret_cast<const u32x4*>(bp + kb * 32 + hi * 16);
    bfr.u4[0] = pb[0]; bfr.u4[1] = pb[1];
    acc = __builtin_amdgcn_wmma_f32_16x16x32_bf16(false, a.bf, false, bfr.bf,
                                                  (short)0, acc, false, false);
  }
#pragma unroll
  for (int kb = 0; kb < 8; ++kb) {       // second half: output_one * W2[:, 256:]
    Frag a, bfr;
    const u32x4* pa = reinterpret_cast<const u32x4*>(aop + kb * 32 + hi * 16);
    a.u4[0] = pa[0]; a.u4[1] = pa[1];
    const u32x4* pb = reinterpret_cast<const u32x4*>(bp + D_ + kb * 32 + hi * 16);
    bfr.u4[0] = pb[0]; bfr.u4[1] = pb[1];
    acc = __builtin_amdgcn_wmma_f32_16x16x32_bf16(false, a.bf, false, bfr.bf,
                                                  (short)0, acc, false, false);
  }
  const int col = ct * 16 + ln;
  const float bias = b2[col];
#pragma unroll
  for (int v = 0; v < 8; ++v) {
    int row = rt * 16 + v + hi * 8;
    float z = acc[v] + bias;
    float sg = 1.0f / (1.0f + __expf(-z));
    out[(size_t)row * O_ + col] = sg * tanhf(z);
  }
}

// ---------------------------------------------------------------------------
extern "C" void kernel_launch(void* const* d_in, const int* in_sizes, int n_in,
                              void* d_out, int out_size, void* d_ws, size_t ws_size,
                              hipStream_t stream) {
  const float* input  = (const float*)d_in[0];
  const float* memory = (const float*)d_in[1];
  const float* mask   = (const float*)d_in[2];
  const float* Win1   = (const float*)d_in[3];
  const float* bin1   = (const float*)d_in[4];
  const float* Wmem1  = (const float*)d_in[5];
  const float* bmem1  = (const float*)d_in[6];
  const float* W2     = (const float*)d_in[7];
  const float* b2     = (const float*)d_in[8];
  float* out = (float*)d_out;

  // workspace: six 4M-elem bf16 tensors + bf16 weights  (~49 MB)
  unsigned short* ws = (unsigned short*)d_ws;
  unsigned short* Qd = ws;                               // [B*Ld, H]
  unsigned short* Kd = Qd + (size_t)B_ * LD_ * H_;       // [B*Lm, H]
  unsigned short* Vt = Kd + (size_t)B_ * LM_ * H_;       // [B, D, Lm]
  unsigned short* Xb = Vt + (size_t)B_ * LM_ * D_;       // [B*Ld, D]
  unsigned short* O1 = Xb + (size_t)B_ * LD_ * D_;       // [B*Ld, D]
  unsigned short* Mb = O1 + (size_t)B_ * LD_ * D_;       // [B*Lm, D]
  unsigned short* Wbi = Mb + (size_t)B_ * LM_ * D_;      // [H, D]
  unsigned short* Wbm = Wbi + (size_t)H_ * D_;           // [H, D]
  unsigned short* Wb2 = Wbm + (size_t)H_ * D_;           // [O, 2D]

  k_convert<<<2048, 256, 0, stream>>>(input, memory, Win1, Wmem1, W2,
                                      Xb, Mb, Vt, Wbi, Wbm, Wb2);
  k_project<<<4096, 256, 0, stream>>>(Xb, Mb, Wbi, bin1, Wbm, bmem1, Qd, Kd);
  k_attn   <<<128,  256, 0, stream>>>(Qd, Kd, Vt, mask, O1);
  k_glu    <<<2048, 256, 0, stream>>>(Xb, O1, Wb2, b2, out);
}
